// MemInt_49959059587241
// MI455X (gfx1250) — matrix operand order repaired and checked by reading, not compiled, not run on hardware
//
#include <hip/hip_runtime.h>

// ---------------------------------------------------------------------------
// MemInt on MI455X (gfx1250):
//   k = exp(-x)                               [4096 x 4096]
//   C = T @ k,  T[j,l] = h[j-l] (l<=j)        causal Toeplitz GEMM (4096^3)
//   out = dt*(0.5*k0*h_j - 0.5*k_ji*h0 - C) + fe_j
//
// v3: split-precision f16 WMMA (hi/lo, 3 matmuls ~ fp32 accuracy)
//  - planar khi/klo + precomputed Toeplitz tile table (all L2-resident)
//  - double-buffered LDS fed by global_load_async_to_lds_b128 (ASYNCcnt)
//  - strength-reduced staging: global offsets advance by +BK*M_T (B) and
//    -ATILE (A, since qi = 4*by - kt + 3 decrements) per K-tile; LDS targets
//    are base + buf*BUFSTRIDE. No per-iteration address rebuilds.
//  - causality: row-block `by` runs (by+1)*4 K-tiles
// ---------------------------------------------------------------------------

typedef __attribute__((ext_vector_type(16))) _Float16 v16h;
typedef __attribute__((ext_vector_type(8)))  _Float16 v8h;
typedef __attribute__((ext_vector_type(8)))  float    v8f;

#define N_T 4096
#define M_T 4096
#define DT_STEP 0.01f

#define BM 128
#define BN 128
#define BK 32

// LDS row strides (halves). APAD=40: rows 16B-aligned, 20-DW lane stride ->
// 16 distinct banks over the 16 M-lanes. BPAD=136: 272B rows (68 DW = 4*17)
// -> conflict-free lanes 0..15, lanes 16..31 read the K+8 rows.
#define APAD 40
#define BPAD 136

#define NQ 128           // distinct Toeplitz offsets: q = 4*by - kt in [-3,124]
#define ATILE (BM * BK)  // halves per precomputed A tile (unpadded, stride BK)

#define ABUFSTRIDE (BM * APAD * 2)   // bytes between A double buffers
#define BBUFSTRIDE (BK * BPAD * 2)   // bytes between B double buffers

// ---------------------------------------------------------------------------
// gfx1250 async copy: global -> LDS, 16B per lane, tracked by ASYNCcnt.
// ---------------------------------------------------------------------------
__device__ __forceinline__ void async_b128(const _Float16* g, unsigned lds_off) {
    asm volatile("global_load_async_to_lds_b128 %0, %1, off"
                 :: "v"(lds_off), "v"(g) : "memory");
}
__device__ __forceinline__ void wait_async0() {
    asm volatile("s_wait_asynccnt 0x0" ::: "memory");
}
__device__ __forceinline__ unsigned lds_addr(const void* p) {
    return (unsigned)(uintptr_t)p;   // low 32 bits of flat LDS aperture addr
}

// Two aligned 16B LDS loads -> one v16h fragment.
__device__ __forceinline__ v16h frag_cat(const _Float16* p0, const _Float16* p1) {
    v8h a = *(const v8h*)p0;
    v8h b = *(const v8h*)p1;
    return __builtin_shufflevector(a, b, 0,1,2,3,4,5,6,7,8,9,10,11,12,13,14,15);
}

// ---------------------------------------------------------------------------
// Kernel 1: planar hi/lo split of k = exp(-x).
// ---------------------------------------------------------------------------
__global__ void memint_pack_kernel(const float* __restrict__ x,
                                   _Float16* __restrict__ khi,
                                   _Float16* __restrict__ klo) {
    size_t idx = (size_t)blockIdx.x * blockDim.x + threadIdx.x;
    if (idx < (size_t)N_T * M_T) {
        float k = __expf(-x[idx]);
        _Float16 hi = (_Float16)k;
        khi[idx] = hi;
        klo[idx] = (_Float16)(k - (float)hi);
    }
}

// ---------------------------------------------------------------------------
// Kernel 2: precompute all 128 distinct Toeplitz A tiles (hi/lo planar).
// Tile qi holds A[j][l] = h[32*(qi-3) + j - l] (causal, 0 above diagonal),
// stored [128][32] row-major so a K-tile copy is a contiguous 8KB memcpy.
// ---------------------------------------------------------------------------
__global__ void memint_atab_kernel(const float* __restrict__ h,
                                   _Float16* __restrict__ Athi,
                                   _Float16* __restrict__ Atlo) {
    const int qi  = blockIdx.x;          // 0..127
    const int t   = threadIdx.x;
    const int row = t >> 1;
    const int cc  = (t & 1) * 16;
    const int dbase = 32 * (qi - 3) + row - cc;
    #pragma unroll
    for (int c = 0; c < 16; ++c) {
        int d = dbase - c;
        float v = (d >= 0) ? h[d] : 0.0f;
        _Float16 hi = (_Float16)v;
        size_t o = (size_t)qi * ATILE + row * BK + cc + c;
        Athi[o] = hi;
        Atlo[o] = (_Float16)(v - (float)hi);
    }
}

// ---------------------------------------------------------------------------
// Kernel 3: causal split-f16 WMMA GEMM + fused epilogue.
// Grid (M_T/BN, N_T/BM), block 256 = 8 wave32 (4 M x 2 N), wave tile 32x64.
// ---------------------------------------------------------------------------
__global__ void __launch_bounds__(256, 2)
memint_gemm_kernel(const _Float16* __restrict__ khi,
                   const _Float16* __restrict__ klo,
                   const _Float16* __restrict__ Athi,
                   const _Float16* __restrict__ Atlo,
                   const float* __restrict__ h,
                   const float* __restrict__ fe,
                   float* __restrict__ out) {
    __shared__ _Float16 Ahi[2][BM][APAD];
    __shared__ _Float16 Alo[2][BM][APAD];
    __shared__ _Float16 Bhi[2][BK][BPAD];
    __shared__ _Float16 Blo[2][BK][BPAD];

    const int t    = threadIdx.x;
    const int lane = t & 31;
    const int wave = t >> 5;
    const int m0   = (wave >> 1) * 32;
    const int n0   = (wave & 1) * 64;

    const int by = blockIdx.y;
    const int j0 = by * BM;
    const int cb = blockIdx.x * BN;

    const int ktiles = (by + 1) * (BM / BK);   // causality

    v8f acc[2][4];
    #pragma unroll
    for (int mt = 0; mt < 2; ++mt)
        #pragma unroll
        for (int nt = 0; nt < 4; ++nt)
            acc[mt][nt] = (v8f){0.f,0.f,0.f,0.f,0.f,0.f,0.f,0.f};

    // ---- strength-reduced staging state: per-thread global pointers that
    // advance by constants per K-tile, and fixed buffer-0 LDS targets.
    const _Float16* gB[2];     // into khi; klo shares the same offset
    const _Float16* gA[2];     // into Athi; Atlo shares the same offset
    unsigned lB[2], lBl[2], lA[2], lAl[2];
    {
        #pragma unroll
        for (int i = 0; i < 2; ++i) {
            const int c  = t + i * 256;
            const int br = c >> 4, bc = (c & 15) * 8;       // B: 32 x (16x8)
            const int ar = c >> 2, ac = (c & 3) * 8;        // A: 128 x (4x8)
            gB[i]  = khi + (size_t)br * M_T + cb + bc;
            gA[i]  = Athi + (size_t)(4 * by + 3) * ATILE + ar * BK + ac;
            lB[i]  = lds_addr(&Bhi[0][br][bc]);
            lBl[i] = lds_addr(&Blo[0][br][bc]);
            lA[i]  = lds_addr(&Ahi[0][ar][ac]);
            lAl[i] = lds_addr(&Alo[0][ar][ac]);
        }
    }
    const size_t    kplane = (size_t)N_T * M_T;      // khi -> klo delta
    const ptrdiff_t aplane = (ptrdiff_t)NQ * ATILE;  // Athi -> Atlo delta

    auto stage = [&](int buf) {   // issues tile, then bumps pointers
        const unsigned bB = (unsigned)buf * BBUFSTRIDE;
        const unsigned bA = (unsigned)buf * ABUFSTRIDE;
        #pragma unroll
        for (int i = 0; i < 2; ++i) {
            async_b128(gB[i],          lB[i]  + bB);
            async_b128(gB[i] + kplane, lBl[i] + bB);
            async_b128(gA[i],          lA[i]  + bA);
            async_b128(gA[i] + aplane, lAl[i] + bA);
            gB[i] += (size_t)BK * M_T;   // next K-tile rows of k
            gA[i] -= ATILE;              // qi decrements with kt
        }
    };

    stage(0);   // tile 0 -> buffer 0

    const int kb = (lane >> 4) * 8;   // A fragment K-chunk base (ISA 7.12.2)

    for (int kt = 0; kt < ktiles; ++kt) {
        const int buf = kt & 1;
        wait_async0();        // tile kt landed in LDS (only tile in flight)
        __syncthreads();      // visible to all waves; prev reads of buf^1 done

        if (kt + 1 < ktiles) stage(buf ^ 1);   // copy kt+1 under compute kt

        // ---- fragments
        v16h ahi[2], alo[2], bhi[4], blo[4];
        #pragma unroll
        for (int mt = 0; mt < 2; ++mt) {
            const _Float16* rh = &Ahi[buf][m0 + mt * 16 + (lane & 15)][0];
            const _Float16* rl = &Alo[buf][m0 + mt * 16 + (lane & 15)][0];
            ahi[mt] = frag_cat(rh + kb, rh + kb + 16);
            alo[mt] = frag_cat(rl + kb, rl + kb + 16);
        }
        #pragma unroll
        for (int nt = 0; nt < 4; ++nt) {
            const int nn = n0 + nt * 16 + (lane & 15);
            v16h vh, vl;
            #pragma unroll
            for (int s = 0; s < 16; ++s) {
                const int kk = kb + (s & 7) + ((s >> 3) << 4);
                vh[s] = Bhi[buf][kk][nn];
                vl[s] = Blo[buf][kk][nn];
            }
            bhi[nt] = vh;
            blo[nt] = vl;
        }

        // ---- C += Ahi*Bhi + Ahi*Blo + Alo*Bhi   (drop Alo*Blo ~ 2^-22)
        #pragma unroll
        for (int mt = 0; mt < 2; ++mt) {
            #pragma unroll
            for (int nt = 0; nt < 4; ++nt) {
                acc[mt][nt] = __builtin_amdgcn_wmma_f32_16x16x32_f16(
                    false, ahi[mt], false, bhi[nt], (short)0, acc[mt][nt], false, false);
                acc[mt][nt] = __builtin_amdgcn_wmma_f32_16x16x32_f16(
                    false, ahi[mt], false, blo[nt], (short)0, acc[mt][nt], false, false);
                acc[mt][nt] = __builtin_amdgcn_wmma_f32_16x16x32_f16(
                    false, alo[mt], false, bhi[nt], (short)0, acc[mt][nt], false, false);
            }
        }
    }

    // ---- epilogue: out = dt*(0.5*k0*h_j - 0.5*k_ji*h0 - C) + fe_j
    // C/D layout: VGPR r -> M = r + 8*(lane>=16); lane&15 -> N.
    const float h0 = h[0];
    #pragma unroll
    for (int nt = 0; nt < 4; ++nt) {
        const int ig = cb + n0 + nt * 16 + (lane & 15);
        const float k0 = (float)khi[ig] + (float)klo[ig];   // row 0 of k
        #pragma unroll
        for (int mt = 0; mt < 2; ++mt) {
            const int jb = j0 + m0 + mt * 16 + ((lane >> 4) << 3);
            #pragma unroll
            for (int r = 0; r < 8; ++r) {
                const int jg = jb + r;
                const size_t o = (size_t)jg * M_T + ig;
                const float kji = (float)khi[o] + (float)klo[o];
                out[o] = DT_STEP * (0.5f * k0 * h[jg] - 0.5f * kji * h0 - acc[mt][nt][r])
                         + fe[jg];
            }
        }
    }
}

// ---------------------------------------------------------------------------
extern "C" void kernel_launch(void* const* d_in, const int* in_sizes, int n_in,
                              void* d_out, int out_size, void* d_ws, size_t ws_size,
                              hipStream_t stream) {
    const float* x  = (const float*)d_in[0];   // [1, 4096, 4096]
    const float* fe = (const float*)d_in[1];   // [1, 4096]
    const float* h  = (const float*)d_in[2];   // [1, 4096]
    float* out = (float*)d_out;

    // workspace: khi (32MB) | klo (32MB) | Atab_hi (1MB) | Atab_lo (1MB)
    _Float16* khi  = (_Float16*)d_ws;
    _Float16* klo  = khi + (size_t)N_T * M_T;
    _Float16* Athi = klo + (size_t)N_T * M_T;
    _Float16* Atlo = Athi + (size_t)NQ * ATILE;

    {   // Phase 1: planar hi/lo split of k = exp(-x)
        const int threads = 256;
        const int blocks  = (N_T * M_T + threads - 1) / threads;
        memint_pack_kernel<<<blocks, threads, 0, stream>>>(x, khi, klo);
    }
    {   // Phase 2: 128 Toeplitz A tiles (one-time gather)
        memint_atab_kernel<<<NQ, 256, 0, stream>>>(h, Athi, Atlo);
    }
    {   // Phase 3: causal WMMA GEMM + epilogue
        dim3 grid(M_T / BN, N_T / BM);   // (32, 32)
        memint_gemm_kernel<<<grid, dim3(256), 0, stream>>>(
            khi, klo, Athi, Atlo, h, fe, out);
    }
}
